// layer_34522947125929
// MI455X (gfx1250) — matrix operand order repaired and checked by reading, not compiled, not run on hardware
//
#include <hip/hip_runtime.h>
#include <hip/hip_bf16.h>

// LSTM forward: B=32, T=1024, D_IN=512, H=512.
// Strategy:
//   1) pack_weights/pack_bias: W*,U* (f32, k-major rows) -> f16 "n-major rows,
//      k-contiguous" layout so WMMA B fragments are contiguous 32B/lane loads.
//   2) xw_gemm: whole-chip parallel WMMA GEMM Z[t][b][4*512] = X@[Wi|Wf|Wc|Wo]
//      (hoists half the FLOPs off the serial chain). Used only if ws fits Z.
//   3) lstm_rec: 2 persistent workgroups (16 batch rows each, 1024 thr = 32
//      waves), loop T steps; h in LDS (f16), c in VGPRs, per-step sync is a
//      workgroup barrier. h@U via v_wmma_f32_16x16x32_f16, K=512 -> 16 wmma
//      per gate tile.

typedef __attribute__((ext_vector_type(16))) _Float16 v16h;
typedef __attribute__((ext_vector_type(8)))  _Float16 v8h;
typedef __attribute__((ext_vector_type(8)))  float    v8f;

#define T_STEPS 1024
#define BATCH   32
#define DIM     512
#define NGATE   2048  // 4 * 512

__device__ __forceinline__ float sigmoidf_(float x) {
    return 1.0f / (1.0f + __expf(-x));
}
__device__ __forceinline__ float tanhf_(float x) {
    // 1 - 2/(e^{2x}+1); saturates correctly at +-1 for |x| large
    return 1.0f - 2.0f / (__expf(2.0f * x) + 1.0f);
}

// A fragment (16-bit, 16x32): lane (half = lane>>4, m = lane&15):
//   elems 0..7  = K kk*32 + half*16     + 0..7
//   elems 8..15 = K kk*32 + half*16 + 8 + 0..7
__device__ __forceinline__ v16h load_a_frag(const _Float16* row, int kk, int half) {
    const v8h lo = *(const v8h*)(row + kk * 32 + half * 16);
    const v8h hi = *(const v8h*)(row + kk * 32 + half * 16 + 8);
    v16h r;
#pragma unroll
    for (int j = 0; j < 8; ++j) { r[j] = lo[j]; r[8 + j] = hi[j]; }
    return r;
}

// B fragment (16-bit, 32x16): lane (half = lane>>4, n = lane&15):
//   elems 0..15 = K kk*32 + half*16 + 0..15 (contiguous in k-major layout)
__device__ __forceinline__ v16h load_b_frag(const _Float16* colbase, int kk, int half) {
    return *(const v16h*)(colbase + kk * 32 + half * 16);
}

__device__ __forceinline__ v8f wmma_f16(v16h a, v16h b, v8f c) {
    return __builtin_amdgcn_wmma_f32_16x16x32_f16(
        /*neg_a=*/false, a, /*neg_b=*/false, b,
        /*c_mod=*/(short)0, c, /*reuse_a=*/false, /*reuse_b=*/false);
}

// ---------------------------------------------------------------------------
// Pack W/U: dst[(g*512 + n)*512 + k] = src_g[k*512 + n]  (f32 -> f16, k-major)
__global__ void pack_weights(const float* __restrict__ Wi, const float* __restrict__ Ui,
                             const float* __restrict__ Wf, const float* __restrict__ Uf,
                             const float* __restrict__ Wc, const float* __restrict__ Uc,
                             const float* __restrict__ Wo, const float* __restrict__ Uo,
                             _Float16* __restrict__ Wt, _Float16* __restrict__ Ut) {
    unsigned idx = blockIdx.x * blockDim.x + threadIdx.x;   // 2*4*512*512 total
    int k = idx & 511;
    int n = (idx >> 9) & 511;
    int g = (idx >> 18) & 3;
    int which = idx >> 20;                                  // 0 = W, 1 = U
    const float* srcs[8] = {Wi, Wf, Wc, Wo, Ui, Uf, Uc, Uo};
    const float* s = srcs[which * 4 + g];
    _Float16* d = which ? Ut : Wt;
    d[(((size_t)g * 512 + n) << 9) + k] = (_Float16)s[((size_t)k << 9) + n];
}

__global__ void pack_bias(const float* __restrict__ bi, const float* __restrict__ bf,
                          const float* __restrict__ bc, const float* __restrict__ bo,
                          float* __restrict__ bias4) {
    int idx = blockIdx.x * blockDim.x + threadIdx.x;        // 2048
    const float* bs[4] = {bi, bf, bc, bo};
    bias4[idx] = bs[idx >> 9][idx & 511];
}

// ---------------------------------------------------------------------------
// Z[t][b][N] = (X[b,t,:] @ Wt_col(N)),  N = g*512+n.  Fully parallel WMMA GEMM.
// grid = (16, 2, 1024), block = 256 (8 waves, 1 N-tile each)
__global__ void __launch_bounds__(256) xw_gemm(const float* __restrict__ X,
                                               const _Float16* __restrict__ Wt,
                                               _Float16* __restrict__ Z) {
    __shared__ _Float16 xs[16][DIM];     // 16 rows of x_t, f16
    const int t = blockIdx.z;
    const int b0 = blockIdx.y * 16;

    for (int i = threadIdx.x; i < 16 * DIM; i += 256) {
        int m = i >> 9, k = i & 511;
        xs[m][k] = (_Float16)X[((size_t)(b0 + m) * T_STEPS + t) * DIM + k];
    }
    __syncthreads();

    const int lane = threadIdx.x & 31;
    const int wave = threadIdx.x >> 5;
    const int half = lane >> 4;
    const int ln = lane & 15;
    const int N = blockIdx.x * 128 + wave * 16 + ln;

    const _Float16* wcol = Wt + (size_t)N * DIM;
    v8f acc = {};
#pragma unroll
    for (int kk = 0; kk < 16; ++kk) {
        v16h a = load_a_frag(&xs[ln][0], kk, half);
        v16h b = load_b_frag(wcol, kk, half);
        acc = wmma_f16(a, b, acc);
    }
#pragma unroll
    for (int r = 0; r < 8; ++r) {
        int mm = r + 8 * half;
        Z[((size_t)t * BATCH + b0 + mm) * NGATE + N] = (_Float16)acc[r];
    }
}

// ---------------------------------------------------------------------------
// Persistent recurrent kernel. grid = 2 blocks x 1024 threads (32 waves).
// Block owns batch rows [b0, b0+16). Wave w owns h-columns [w*16, w*16+16)
// across all 4 gates; c tile lives in its VGPRs for the whole T loop.
__global__ void __launch_bounds__(1024) lstm_rec(const float* __restrict__ X,
                                                 const float* __restrict__ init,
                                                 const _Float16* __restrict__ Wt,
                                                 const _Float16* __restrict__ Ut,
                                                 const float* __restrict__ bias4,
                                                 const _Float16* __restrict__ Z,
                                                 float* __restrict__ out,
                                                 int useZ) {
    __shared__ _Float16 hbuf[16][DIM];   // current hidden state, f16 (16 KB)
    __shared__ _Float16 xbuf[16][DIM];   // staged x_t, f16 (16 KB; fused path)

    const int b0 = blockIdx.x * 16;
    const int lane = threadIdx.x & 31;
    const int wave = threadIdx.x >> 5;
    const int half = lane >> 4;
    const int ln = lane & 15;
    const int n0 = wave * 16;
    const int n = n0 + ln;

    // h0 -> LDS (f16)
    for (int i = threadIdx.x; i < 16 * DIM; i += 1024) {
        int m = i >> 9, k = i & 511;
        hbuf[m][k] = (_Float16)init[(size_t)(b0 + m) * DIM + k];
    }
    // c0 -> registers (C-matrix layout: reg r -> row r + 8*half, col n)
    v8f c;
#pragma unroll
    for (int r = 0; r < 8; ++r) {
        int m = r + 8 * half;
        c[r] = init[(size_t)(BATCH + b0 + m) * DIM + n];
    }
    float bval[4];
#pragma unroll
    for (int g = 0; g < 4; ++g) bval[g] = bias4[g * DIM + n];

    __syncthreads();

    for (int t = 0; t < T_STEPS; ++t) {
        if (!useZ) {
            for (int i = threadIdx.x; i < 16 * DIM; i += 1024) {
                int m = i >> 9, k = i & 511;
                xbuf[m][k] = (_Float16)X[((size_t)(b0 + m) * T_STEPS + t) * DIM + k];
            }
        }
        __syncthreads();   // x staged; previous step's h writes visible

        v8f acc[4];
        if (useZ) {
            const _Float16* zrow = Z + ((size_t)t * BATCH + b0) * NGATE;
#pragma unroll
            for (int g = 0; g < 4; ++g)
#pragma unroll
                for (int r = 0; r < 8; ++r)
                    acc[g][r] = bval[g] +
                        (float)zrow[(size_t)(r + 8 * half) * NGATE + g * DIM + n];
        } else {
#pragma unroll
            for (int g = 0; g < 4; ++g)
#pragma unroll
                for (int r = 0; r < 8; ++r) acc[g][r] = bval[g];
            // x_t @ W (K = 512)
#pragma unroll 4
            for (int kk = 0; kk < 16; ++kk) {
                v16h a = load_a_frag(&xbuf[ln][0], kk, half);
#pragma unroll
                for (int g = 0; g < 4; ++g) {
                    const _Float16* col = Wt + ((size_t)g * 512 + n) * DIM;
                    v16h b = load_b_frag(col, kk, half);
                    acc[g] = wmma_f16(a, b, acc[g]);
                }
            }
        }

        // h @ U (K = 512) — the serial critical path
#pragma unroll 4
        for (int kk = 0; kk < 16; ++kk) {
            v16h a = load_a_frag(&hbuf[ln][0], kk, half);
#pragma unroll
            for (int g = 0; g < 4; ++g) {
                const _Float16* col = Ut + ((size_t)g * 512 + n) * DIM;
                v16h b = load_b_frag(col, kk, half);
                acc[g] = wmma_f16(a, b, acc[g]);
            }
        }

        __syncthreads();   // all waves finished reading hbuf for this step

        // gates + state update; write h_new to LDS (next step) and d_out
#pragma unroll
        for (int r = 0; r < 8; ++r) {
            int m = r + 8 * half;
            float iv = sigmoidf_(acc[0][r]);
            float fv = sigmoidf_(acc[1][r]);
            float gv = tanhf_(acc[2][r]);
            float ov = sigmoidf_(acc[3][r]);
            float cn = fv * c[r] + iv * gv;
            c[r] = cn;
            float hn = ov * tanhf_(cn);
            out[((size_t)(b0 + m) * T_STEPS + t) * DIM + n] = hn;
            hbuf[m][n] = (_Float16)hn;
        }
    }
}

// ---------------------------------------------------------------------------
extern "C" void kernel_launch(void* const* d_in, const int* in_sizes, int n_in,
                              void* d_out, int out_size, void* d_ws, size_t ws_size,
                              hipStream_t stream) {
    const float* X    = (const float*)d_in[0];
    const float* init = (const float*)d_in[1];
    const float* Wi = (const float*)d_in[2],  *Ui = (const float*)d_in[3],  *bi = (const float*)d_in[4];
    const float* Wf = (const float*)d_in[5],  *Uf = (const float*)d_in[6],  *bf = (const float*)d_in[7];
    const float* Wc = (const float*)d_in[8],  *Uc = (const float*)d_in[9],  *bc = (const float*)d_in[10];
    const float* Wo = (const float*)d_in[11], *Uo = (const float*)d_in[12], *bo = (const float*)d_in[13];
    float* out = (float*)d_out;

    char* ws = (char*)d_ws;
    _Float16* Wt   = (_Float16*)ws;                              // 2 MiB
    _Float16* Ut   = (_Float16*)(ws + (2u << 20));               // 2 MiB
    float*    bias = (float*)   (ws + (4u << 20));               // 8 KiB
    _Float16* Z    = (_Float16*)(ws + (4u << 20) + 65536);       // 128 MiB
    const size_t zbytes = (size_t)T_STEPS * BATCH * NGATE * sizeof(_Float16);
    const int useZ = (ws_size >= (4u << 20) + 65536 + zbytes) ? 1 : 0;

    pack_weights<<<8192, 256, 0, stream>>>(Wi, Ui, Wf, Uf, Wc, Uc, Wo, Uo, Wt, Ut);
    pack_bias<<<8, 256, 0, stream>>>(bi, bf, bc, bo, bias);
    if (useZ) {
        dim3 grid(16, 2, 1024);
        xw_gemm<<<grid, 256, 0, stream>>>(X, Wt, Z);
    }
    lstm_rec<<<2, 1024, 0, stream>>>(X, init, Wt, Ut, bias, useZ ? Z : (const _Float16*)nullptr,
                                     out, useZ);
}
